// GATNodeClassifier_34299608826092
// MI455X (gfx1250) — matrix operand order repaired and checked by reading, not compiled, not run on hardware
//
#include <hip/hip_runtime.h>
#include <hip/hip_bf16.h>

// ---------------------------------------------------------------------------
// Types for CDNA5 WMMA (wave32)
// ---------------------------------------------------------------------------
typedef __attribute__((ext_vector_type(16))) __bf16       v16bf;
typedef __attribute__((ext_vector_type(8)))  float        v8f;
typedef __attribute__((ext_vector_type(8)))  unsigned int u32x8;

#define F_IN    128
#define HEADS1  4
#define HID1    32
#define F_HID   128   /* HEADS1*HID1 */
#define OUT_C   8
#define LD_H2   16    /* padded leading dim of layer-2 GEMM output */
#define NEG_SLOPE 0.2f

static inline int cdiv_i(int a, int b) { return (a + b - 1) / b; }

// ---------------------------------------------------------------------------
// Ordered-uint encoding for float atomic max (monotonic map R -> u32)
// ---------------------------------------------------------------------------
__device__ __forceinline__ unsigned enc_f32(float f) {
  unsigned b = __float_as_uint(f);
  return (b & 0x80000000u) ? ~b : (b | 0x80000000u);
}
__device__ __forceinline__ float dec_f32(unsigned u) {
  return __uint_as_float((u & 0x80000000u) ? (u & 0x7FFFFFFFu) : ~u);
}

// ---------------------------------------------------------------------------
// Elementwise conversions
// ---------------------------------------------------------------------------
__global__ void f32_to_bf16_kernel(const float* __restrict__ in,
                                   __bf16* __restrict__ out, int n) {
  int t = blockIdx.x * blockDim.x + threadIdx.x;
  if (t < n) out[t] = (__bf16)in[t];
}

// W[K][Ncol] row-major  ->  Wt[NcolPad][K] bf16 (column-major B for WMMA)
__global__ void transpose_pad_bf16_kernel(const float* __restrict__ W,
                                          __bf16* __restrict__ Wt,
                                          int K, int Ncol, int NcolPad) {
  int t = blockIdx.x * blockDim.x + threadIdx.x;
  if (t >= NcolPad * K) return;
  int n = t / K, k = t - n * K;
  Wt[(size_t)n * K + k] = (n < Ncol) ? (__bf16)W[(size_t)k * Ncol + n] : (__bf16)0.0f;
}

// out = bf16(elu(in)), alpha=1
__global__ void elu_to_bf16_kernel(const float* __restrict__ in,
                                   __bf16* __restrict__ out, int n) {
  int t = blockIdx.x * blockDim.x + threadIdx.x;
  if (t >= n) return;
  float v = in[t];
  v = (v > 0.0f) ? v : (__expf(v) - 1.0f);
  out[t] = (__bf16)v;
}

// ---------------------------------------------------------------------------
// WMMA bf16 GEMM:  C[M x (Ntiles*16)] = A[M x K] * Bt[(Ntiles*16) x K]^T
// A row-major bf16, Bt holds B column-major (row n = column n of B, length K).
// One 16x16 tile per wave32, 4 waves per block, K stepped by 32.
// ---------------------------------------------------------------------------
__device__ __forceinline__ v16bf load_a_frag(const __bf16* p) {
  // K = {0..15} contiguous relative to p (k = 16*half + s order, same as the B fragment)
  uint4 lo = *reinterpret_cast<const uint4*>(p);
  uint4 hi = *reinterpret_cast<const uint4*>(p + 8);
  u32x8 u = {lo.x, lo.y, lo.z, lo.w, hi.x, hi.y, hi.z, hi.w};
  return __builtin_bit_cast(v16bf, u);
}
__device__ __forceinline__ v16bf load_b_frag(const __bf16* p) {
  // 16 contiguous K values per lane (k = 16*half + s order, same as the A fragment)
  u32x8 u = *reinterpret_cast<const u32x8*>(p);
  return __builtin_bit_cast(v16bf, u);
}

__global__ void __launch_bounds__(128)
wmma_gemm_bf16_kernel(const __bf16* __restrict__ A, const __bf16* __restrict__ Bt,
                      float* __restrict__ C, int M, int Mtiles, int Ntiles,
                      int K, int ldc) {
  const int wave = threadIdx.x >> 5;
  const int lane = threadIdx.x & 31;
  const int tile = blockIdx.x * 4 + wave;
  if (tile >= Mtiles * Ntiles) return;            // whole-wave uniform exit
  const int tm = tile / Ntiles;
  const int tn = tile - tm * Ntiles;
  const int half = lane >> 4;
  const int l15  = lane & 15;

  int arow = tm * 16 + l15;
  if (arow >= M) arow = M - 1;                    // clamp loads for ragged M
  const __bf16* ap = A  + (size_t)arow * K;
  const __bf16* bp = Bt + (size_t)(tn * 16 + l15) * K;

  v8f acc = {};
  for (int k0 = 0; k0 < K; k0 += 32) {
    v16bf a = load_a_frag(ap + k0 + half * 16);
    v16bf b = load_b_frag(bp + k0 + half * 16);
    acc = __builtin_amdgcn_wmma_f32_16x16x32_bf16(
        /*neg_a=*/false, a, /*neg_b=*/false, b,
        /*c_mod=*/(short)0, acc, /*reuse_a=*/false, /*reuse_b=*/false);
  }
#pragma unroll
  for (int r = 0; r < 8; ++r) {
    int row = tm * 16 + r + half * 8;
    if (row < M) C[(size_t)row * ldc + tn * 16 + l15] = acc[r];
  }
}

// ---------------------------------------------------------------------------
// Per-node attention coefficients: as[n,h] = <H[n,h,:], a_src[h,:]>, same dst
// ---------------------------------------------------------------------------
template <int HEADS, int C, int LDH>
__global__ void node_alpha_kernel(const float* __restrict__ H,
                                  const float* __restrict__ a_src,
                                  const float* __restrict__ a_dst,
                                  float* __restrict__ as, float* __restrict__ ad,
                                  int N) {
  int t = blockIdx.x * blockDim.x + threadIdx.x;
  if (t >= N * HEADS) return;
  int n = t / HEADS, h = t - n * HEADS;
  const float* hp = H + (size_t)n * LDH + h * C;
  float s = 0.0f, d = 0.0f;
#pragma unroll
  for (int c = 0; c < C; ++c) {
    float v = hp[c];
    s += v * a_src[h * C + c];
    d += v * a_dst[h * C + c];
  }
  as[t] = s;
  ad[t] = d;
}

// out[n,f] = bias[f]; segmax=0 (encoded -inf floor), segsum=0
template <int H, int F>
__global__ void init_layer_kernel(unsigned* __restrict__ segmax,
                                  float* __restrict__ segsum,
                                  float* __restrict__ out,
                                  const float* __restrict__ bias,
                                  int N) {
  int t = blockIdx.x * blockDim.x + threadIdx.x;
  if (t >= N * F) return;
  out[t] = bias[t % F];
  if (t < N * H) { segmax[t] = 0u; segsum[t] = 0.0f; }
}

// ---------------------------------------------------------------------------
// Edge passes (edges e<E from edge_index; e>=E are self-loops)
// ---------------------------------------------------------------------------
template <int H>
__global__ void edge_logits_max_kernel(const int* __restrict__ ei, int E, int ET,
                                       const float* __restrict__ as,
                                       const float* __restrict__ ad,
                                       float* __restrict__ logits,
                                       unsigned* __restrict__ segmax) {
  int t = blockIdx.x * blockDim.x + threadIdx.x;
  if (t >= ET * H) return;
  int e = t / H, h = t - e * H;
  int src = (e < E) ? ei[e]     : (e - E);
  int dst = (e < E) ? ei[E + e] : (e - E);
  float lg = as[src * H + h] + ad[dst * H + h];
  lg = (lg > 0.0f) ? lg : NEG_SLOPE * lg;         // leaky_relu
  logits[t] = lg;
  atomicMax(&segmax[dst * H + h], enc_f32(lg));
}

template <int H>
__global__ void edge_exp_sum_kernel(const int* __restrict__ ei, int E, int ET,
                                    float* __restrict__ logits /* in: logit, out: e */,
                                    const unsigned* __restrict__ segmax,
                                    float* __restrict__ segsum) {
  int t = blockIdx.x * blockDim.x + threadIdx.x;
  if (t >= ET * H) return;
  int e = t / H, h = t - e * H;
  int dst = (e < E) ? ei[E + e] : (e - E);
  float m  = dec_f32(segmax[dst * H + h]);
  float ev = __expf(logits[t] - m);
  logits[t] = ev;
  atomicAdd(&segsum[dst * H + h], ev);
}

// One thread per (edge, feature): wave of 32 lanes covers 32 consecutive
// channels of one src row (coalesced 128B gather) and 32 consecutive
// channels of one dst row (cacheline-granular L2 atomic scatter).
template <int H, int C, int LDH, int LDO>
__global__ void edge_scatter_kernel(const int* __restrict__ ei, int E, int ET,
                                    const float* __restrict__ evals,
                                    const float* __restrict__ segsum,
                                    const float* __restrict__ Hfeat,
                                    float* __restrict__ out) {
  constexpr int HC = H * C;
  long long t = (long long)blockIdx.x * blockDim.x + threadIdx.x;
  if (t >= (long long)ET * HC) return;
  int e = (int)(t / HC);
  int f = (int)(t - (long long)e * HC);           // f = h*C + c
  int h = f / C;
  int src = (e < E) ? ei[e]     : (e - E);
  int dst = (e < E) ? ei[E + e] : (e - E);
  float alpha = evals[e * H + h] / segsum[dst * H + h];  // broadcast loads
  float v = Hfeat[(size_t)src * LDH + f] * alpha;        // coalesced gather
  atomicAdd(&out[(size_t)dst * LDO + f], v);             // coalesced atomics
}

// ---------------------------------------------------------------------------
// Host-side orchestration
// ---------------------------------------------------------------------------
extern "C" void kernel_launch(void* const* d_in, const int* in_sizes, int n_in,
                              void* d_out, int out_size, void* d_ws, size_t ws_size,
                              hipStream_t stream) {
  const float* x        = (const float*)d_in[0];
  const int*   ei       = (const int*)  d_in[1];
  const float* W1       = (const float*)d_in[2];
  const float* a_src1   = (const float*)d_in[3];
  const float* a_dst1   = (const float*)d_in[4];
  const float* b1       = (const float*)d_in[5];
  const float* W2       = (const float*)d_in[6];
  const float* a_src2   = (const float*)d_in[7];
  const float* a_dst2   = (const float*)d_in[8];
  const float* b2       = (const float*)d_in[9];
  float*       out      = (float*)d_out;

  const int N  = in_sizes[0] / F_IN;
  const int E  = in_sizes[1] / 2;
  const int ET = E + N;                       // with self-loops
  const int Mtiles = cdiv_i(N, 16);

  // ---- workspace partition (256B aligned) ----
  char* p = (char*)d_ws;
  auto alloc = [&](size_t bytes) -> void* {
    void* r = (void*)p;
    p += (bytes + 255) & ~(size_t)255;
    return r;
  };
  __bf16*   Xbf     = (__bf16*)  alloc((size_t)N * F_IN * 2);
  __bf16*   Wt1     = (__bf16*)  alloc((size_t)F_HID * F_IN * 2);
  __bf16*   Wt2     = (__bf16*)  alloc((size_t)16 * F_HID * 2);
  float*    H1      = (float*)   alloc((size_t)N * F_HID * 4);
  float*    as1     = (float*)   alloc((size_t)N * HEADS1 * 4);
  float*    ad1     = (float*)   alloc((size_t)N * HEADS1 * 4);
  unsigned* segmax1 = (unsigned*)alloc((size_t)N * HEADS1 * 4);
  float*    segsum1 = (float*)   alloc((size_t)N * HEADS1 * 4);
  float*    e1      = (float*)   alloc((size_t)ET * HEADS1 * 4);
  float*    out1    = (float*)   alloc((size_t)N * F_HID * 4);
  __bf16*   H1bf    = (__bf16*)  alloc((size_t)N * F_HID * 2);
  float*    H2      = (float*)   alloc((size_t)N * LD_H2 * 4);
  float*    as2     = (float*)   alloc((size_t)N * 4);
  float*    ad2     = (float*)   alloc((size_t)N * 4);
  unsigned* segmax2 = (unsigned*)alloc((size_t)N * 4);
  float*    segsum2 = (float*)   alloc((size_t)N * 4);
  float*    e2      = (float*)   alloc((size_t)ET * 4);
  (void)ws_size; (void)n_in; (void)out_size;

  const int TB = 256;

  // ---- prep: bf16 conversions / weight transposes ----
  f32_to_bf16_kernel<<<cdiv_i(N * F_IN, TB), TB, 0, stream>>>(x, Xbf, N * F_IN);
  transpose_pad_bf16_kernel<<<cdiv_i(F_HID * F_IN, TB), TB, 0, stream>>>(
      W1, Wt1, F_IN, F_HID, F_HID);
  transpose_pad_bf16_kernel<<<cdiv_i(16 * F_HID, TB), TB, 0, stream>>>(
      W2, Wt2, F_HID, OUT_C, 16);

  // ---- layer 1 ----
  {
    int tiles = Mtiles * (F_HID / 16);
    wmma_gemm_bf16_kernel<<<cdiv_i(tiles, 4), 128, 0, stream>>>(
        Xbf, Wt1, H1, N, Mtiles, F_HID / 16, F_IN, F_HID);
  }
  node_alpha_kernel<HEADS1, HID1, F_HID><<<cdiv_i(N * HEADS1, TB), TB, 0, stream>>>(
      H1, a_src1, a_dst1, as1, ad1, N);
  init_layer_kernel<HEADS1, F_HID><<<cdiv_i(N * F_HID, TB), TB, 0, stream>>>(
      segmax1, segsum1, out1, b1, N);
  edge_logits_max_kernel<HEADS1><<<cdiv_i(ET * HEADS1, TB), TB, 0, stream>>>(
      ei, E, ET, as1, ad1, e1, segmax1);
  edge_exp_sum_kernel<HEADS1><<<cdiv_i(ET * HEADS1, TB), TB, 0, stream>>>(
      ei, E, ET, e1, segmax1, segsum1);
  {
    long long work = (long long)ET * F_HID;     // one thread per (edge, feature)
    int blocks = (int)((work + TB - 1) / TB);
    edge_scatter_kernel<HEADS1, HID1, F_HID, F_HID><<<blocks, TB, 0, stream>>>(
        ei, E, ET, e1, segsum1, H1, out1);
  }
  elu_to_bf16_kernel<<<cdiv_i(N * F_HID, TB), TB, 0, stream>>>(
      out1, H1bf, N * F_HID);

  // ---- layer 2 (8 cols padded to one 16-wide tile) ----
  {
    int tiles = Mtiles * 1;
    wmma_gemm_bf16_kernel<<<cdiv_i(tiles, 4), 128, 0, stream>>>(
        H1bf, Wt2, H2, N, Mtiles, 1, F_HID, LD_H2);
  }
  node_alpha_kernel<1, OUT_C, LD_H2><<<cdiv_i(N, TB), TB, 0, stream>>>(
      H2, a_src2, a_dst2, as2, ad2, N);
  init_layer_kernel<1, OUT_C><<<cdiv_i(N * OUT_C, TB), TB, 0, stream>>>(
      segmax2, segsum2, out, b2, N);
  edge_logits_max_kernel<1><<<cdiv_i(ET, TB), TB, 0, stream>>>(
      ei, E, ET, as2, ad2, e2, segmax2);
  edge_exp_sum_kernel<1><<<cdiv_i(ET, TB), TB, 0, stream>>>(
      ei, E, ET, e2, segmax2, segsum2);
  {
    long long work = (long long)ET * OUT_C;
    int blocks = (int)((work + TB - 1) / TB);
    edge_scatter_kernel<1, OUT_C, LD_H2, OUT_C><<<blocks, TB, 0, stream>>>(
        ei, E, ET, e2, segsum2, H2, out);
  }
}
